// PointNet2Cls_88854283419900
// MI455X (gfx1250) — matrix operand order repaired and checked by reading, not compiled, not run on hardware
//
#include <hip/hip_runtime.h>

typedef _Float16 v16h __attribute__((ext_vector_type(16)));
typedef _Float16 v8h  __attribute__((ext_vector_type(8)));
typedef float    v8f  __attribute__((ext_vector_type(8)));

static inline int cdiv_ll(long long a, long long b){ return (int)((a + b - 1) / b); }

// ---------------------------------------------------------------------------
// Farthest point sampling: one workgroup per batch, dists live in LDS.
// Matches jax scan: emits incoming `farthest` each step (fi[0]=0), argmax
// picks the first maximal index (strict-> per thread + lower-index tiebreak).
// ---------------------------------------------------------------------------
__global__ __launch_bounds__(1024) void fps_kernel(const float* __restrict__ xyz, int N,
                                                   int npoint, float* __restrict__ new_xyz) {
  __shared__ float sd[8192];
  __shared__ float rv[1024];
  __shared__ int   ri[1024];
  __shared__ int   s_far;
  const int b = blockIdx.x;
  const int t = threadIdx.x;
  const int T = blockDim.x;
  const float* px = xyz + (size_t)b * N * 3;
  for (int i = t; i < N; i += T) sd[i] = 1e10f;
  if (t == 0) s_far = 0;
  __syncthreads();
  for (int it = 0; it < npoint; ++it) {
    const int f = s_far;
    const float cx = px[f*3+0], cy = px[f*3+1], cz = px[f*3+2];
    if (t == 0) {
      float* o = new_xyz + ((size_t)b * npoint + it) * 3;
      o[0] = cx; o[1] = cy; o[2] = cz;
    }
    float bv = -1.0f; int bi = 0;
    for (int i = t; i < N; i += T) {
      float dx = px[i*3+0]-cx, dy = px[i*3+1]-cy, dz = px[i*3+2]-cz;
      float d = dx*dx + dy*dy + dz*dz;
      float nd = fminf(sd[i], d);
      sd[i] = nd;
      if (nd > bv) { bv = nd; bi = i; }
    }
    rv[t] = bv; ri[t] = bi;
    __syncthreads();
    for (int s = T >> 1; s > 0; s >>= 1) {
      if (t < s) {
        float ov = rv[t+s]; int oi = ri[t+s];
        if (ov > rv[t] || (ov == rv[t] && oi < ri[t])) { rv[t] = ov; ri[t] = oi; }
      }
      __syncthreads();
    }
    if (t == 0) s_far = ri[0];
    __syncthreads();
  }
}

// ---------------------------------------------------------------------------
// Ball query: one wave32 per center; ordered first-K selection via ballot.
// Matches reference: indices in ascending order, pad with first neighbor.
// ---------------------------------------------------------------------------
__global__ void ballquery_kernel(const float* __restrict__ xyz, const float* __restrict__ centers,
                                 int* __restrict__ idx, int B, int N, int S, int K, float r2) {
  int wid  = (blockIdx.x * blockDim.x + threadIdx.x) >> 5;
  int lane = threadIdx.x & 31;
  if (wid >= B * S) return;
  int b = wid / S, s = wid % S;
  const float* px = xyz + (size_t)b * N * 3;
  const float* c  = centers + ((size_t)b * S + s) * 3;
  float cx = c[0], cy = c[1], cz = c[2];
  int* out = idx + (size_t)wid * K;
  int cnt = 0, firstidx = -1;
  for (int base = 0; base < N && cnt < K; base += 32) {
    int i = base + lane;
    float dx = px[i*3+0]-cx, dy = px[i*3+1]-cy, dz = px[i*3+2]-cz;
    bool ok = (dx*dx + dy*dy + dz*dz) <= r2;
    unsigned m = (unsigned)__ballot(ok);
    if (firstidx < 0 && m) firstidx = base + __builtin_ctz(m);
    if (ok) {
      int pos = cnt + __builtin_popcount(m & ((1u << lane) - 1u));
      if (pos < K) out[pos] = i;
    }
    cnt += __builtin_popcount(m);
  }
  if (firstidx < 0) firstidx = 0;
  for (int slot = cnt + lane; slot < K; slot += 32) out[slot] = firstidx;
}

// ---------------------------------------------------------------------------
// Gather+group: features = [xyz[idx]-center (3), feat[idx] (Cf), zeros pad]
// ---------------------------------------------------------------------------
__global__ void group_kernel(const float* __restrict__ xyz, const float* __restrict__ feat,
                             int Cf, int N, const float* __restrict__ centers,
                             const int* __restrict__ idx, _Float16* __restrict__ X,
                             int S, int K, int Cpad, int total) {
  int tid = blockIdx.x * blockDim.x + threadIdx.x;
  if (tid >= total) return;
  int s = (tid / K) % S;
  int b = tid / (K * S);
  int id = idx[tid];
  const float* c = centers + ((size_t)b * S + s) * 3;
  const float* p = xyz + ((size_t)b * N + id) * 3;
  _Float16* o = X + (size_t)tid * Cpad;
  o[0] = (_Float16)(p[0] - c[0]);
  o[1] = (_Float16)(p[1] - c[1]);
  o[2] = (_Float16)(p[2] - c[2]);
  const float* f = feat + ((size_t)b * N + id) * Cf;
  for (int j = 0; j < Cf; ++j)     o[3 + j] = (_Float16)f[j];
  for (int j = 3 + Cf; j < Cpad; ++j) o[j] = (_Float16)0.0f;
}

// Group-all (SA3): concat [xyz (3), feat (Cf)], no centering, zero pad.
__global__ void groupall_kernel(const float* __restrict__ xyz, const float* __restrict__ feat,
                                int Cf, int Npts, _Float16* __restrict__ X, int Cpad, int total) {
  int tid = blockIdx.x * blockDim.x + threadIdx.x;
  if (tid >= total) return;
  const float* p = xyz + (size_t)tid * 3;
  const float* f = feat + (size_t)tid * Cf;
  _Float16* o = X + (size_t)tid * Cpad;
  o[0] = (_Float16)p[0]; o[1] = (_Float16)p[1]; o[2] = (_Float16)p[2];
  for (int j = 0; j < Cf; ++j)        o[3 + j] = (_Float16)f[j];
  for (int j = 3 + Cf; j < Cpad; ++j) o[j] = (_Float16)0.0f;
}

// Pad & convert weights: Wp[n][k] (Cpad x Kpad, f16) from W[n][k] (Cout x Cin f32)
__global__ void cvt_weight(const float* __restrict__ W, int Cout, int Cin,
                           _Float16* __restrict__ Wp, int Kpad, int Cpad) {
  int tid = blockIdx.x * blockDim.x + threadIdx.x;
  if (tid >= Cpad * Kpad) return;
  int n = tid / Kpad, k = tid % Kpad;
  Wp[tid] = (n < Cout && k < Cin) ? (_Float16)W[(size_t)n * Cin + k] : (_Float16)0.0f;
}

__global__ void zero2_kernel(float* a, float* b, int n) {
  int t = threadIdx.x;
  if (t < n) { a[t] = 0.0f; b[t] = 0.0f; }
}

// ---------------------------------------------------------------------------
// WMMA GEMM: Y[M x Cpad] = A[M x Kpad] * Wp^T, f16 in / f32 accum / f16 out.
// One wave32 per 16x16 tile; BN sum/sum^2 fused via per-column atomics.
// Fragment layouts (contiguous K order k=16*(lane/16)+s; A one 32B run per lane from A[m][k],
// B one 32B contiguous run per lane from W[n][k], C/D row=(lane/16)*8+r).
// ---------------------------------------------------------------------------
__global__ __launch_bounds__(256) void gemm_wmma(const _Float16* __restrict__ A,
                                                 const _Float16* __restrict__ Wp,
                                                 _Float16* __restrict__ Y,
                                                 float* __restrict__ sums, float* __restrict__ sumsq,
                                                 int M, int Kpad, int Cpad) {
  int wave = (blockIdx.x * blockDim.x + threadIdx.x) >> 5;
  int lane = threadIdx.x & 31;
  int ntiles = Cpad >> 4;
  int mtiles = M >> 4;
  if (wave >= mtiles * ntiles) return;           // wave-uniform: EXEC stays all-ones
  int mt = wave / ntiles, nt = wave % ntiles;
  int lr = lane & 15, lh = lane >> 4;
  const _Float16* arow = A  + (size_t)(mt * 16 + lr) * Kpad + lh * 16;
  const _Float16* brow = Wp + (size_t)(nt * 16 + lr) * Kpad + lh * 16;
  v8f acc = {};
  for (int k = 0; k < Kpad; k += 32) {
    v8h a0 = *(const v8h*)(arow + k);            // K = k + lh*16 + 0..7
    v8h a1 = *(const v8h*)(arow + k + 8);        // K = k + lh*16 + 8..15
    v16h b = *(const v16h*)(brow + k);           // K = k + lh*16 + 0..15
    v16h a;
    for (int j = 0; j < 8; ++j) { a[j] = a0[j]; a[j + 8] = a1[j]; }
    acc = __builtin_amdgcn_wmma_f32_16x16x32_f16(false, a, false, b,
                                                 (short)0, acc, false, false);
  }
  int col = nt * 16 + lr;
  float s = 0.0f, s2 = 0.0f;
  for (int r = 0; r < 8; ++r) {
    int row = mt * 16 + lh * 8 + r;
    float v = acc[r];
    Y[(size_t)row * Cpad + col] = (_Float16)v;
    s += v; s2 += v * v;
  }
  atomicAdd(&sums[col],  s);
  atomicAdd(&sumsq[col], s2);
}

// BN (batch stats) + ReLU, in place on f16 Y; pad channels forced to zero.
__global__ void bn_relu(_Float16* __restrict__ Y, const float* __restrict__ sums,
                        const float* __restrict__ sumsq, const float* __restrict__ g,
                        const float* __restrict__ beta, int M, int Cout, int Cpad, long long total) {
  long long tid = (long long)blockIdx.x * blockDim.x + threadIdx.x;
  if (tid >= total) return;
  int c = (int)(tid % Cpad);
  if (c < Cout) {
    float invM = 1.0f / (float)M;
    float mean = sums[c] * invM;
    float var  = sumsq[c] * invM - mean * mean;
    float y = (float)Y[tid];
    float v = (y - mean) * rsqrtf(var + 1e-5f) * g[c] + beta[c];
    Y[tid] = (_Float16)fmaxf(v, 0.0f);
  } else {
    Y[tid] = (_Float16)0.0f;
  }
}

// Max over K group samples: (G*K x Cpad f16) -> (G x C f32)
__global__ void maxpool_kernel(const _Float16* __restrict__ Y, float* __restrict__ out,
                               int G, int K, int C, int Cpad) {
  int tid = blockIdx.x * blockDim.x + threadIdx.x;
  if (tid >= G * C) return;
  int c = tid % C, gi = tid / C;
  float m = -1e30f;
  for (int k = 0; k < K; ++k) {
    float v = (float)Y[((size_t)gi * K + k) * Cpad + c];
    m = fmaxf(m, v);
  }
  out[(size_t)gi * C + c] = m;
}

__global__ void cvt_f16(const float* __restrict__ in, _Float16* __restrict__ out, int n) {
  int tid = blockIdx.x * blockDim.x + threadIdx.x;
  if (tid < n) out[tid] = (_Float16)in[tid];
}

// Final FC (no BN): out[b][o] = sum_k x[b][k]*W[o][k] + bias[o]
__global__ void fc_naive(const _Float16* __restrict__ x, const float* __restrict__ W,
                         const float* __restrict__ bias, float* __restrict__ out,
                         int Bn, int O, int Kin) {
  int tid = blockIdx.x * blockDim.x + threadIdx.x;
  if (tid >= Bn * O) return;
  int o = tid % O, b = tid / O;
  float acc = bias[o];
  for (int k = 0; k < Kin; ++k)
    acc += (float)x[(size_t)b * Kin + k] * W[(size_t)o * Kin + k];
  out[tid] = acc;
}

// ---------------------------------------------------------------------------
extern "C" void kernel_launch(void* const* d_in, const int* in_sizes, int n_in,
                              void* d_out, int out_size, void* d_ws, size_t ws_size,
                              hipStream_t stream) {
  (void)n_in; (void)out_size; (void)ws_size;
  // layers 0..8 = sa1 L1..sa3 L3, 9..11 = fc1..fc3
  const int LC[12] = {64, 64, 128, 128, 128, 256, 256, 256, 512, 256, 128, 40};
  const int LI[12] = {6, 64, 64, 131, 128, 128, 259, 256, 256, 512, 256, 128};
  const float *W[12], *Bb[12], *G[12], *Bt[12];
  const float* xyz;
  if (in_sizes[0] == 16 * 8192 * 3) {
    // insertion-order flatten: xyz, sa1..sa3, fc1..fc3; per-layer W,b,g,beta
    xyz = (const float*)d_in[0];
    int p = 1;
    for (int l = 0; l < 12; ++l) {
      W[l] = (const float*)d_in[p]; Bb[l] = (const float*)d_in[p+1];
      G[l] = (const float*)d_in[p+2]; Bt[l] = (const float*)d_in[p+3]; p += 4;
    }
  } else {
    // sorted-key flatten: fc1,fc2,fc3,sa1..sa3 then xyz; per-layer W,b,beta,g
    static const int ord[12] = {9, 10, 11, 0, 1, 2, 3, 4, 5, 6, 7, 8};
    int p = 0;
    for (int t = 0; t < 12; ++t) {
      int l = ord[t];
      W[l] = (const float*)d_in[p]; Bb[l] = (const float*)d_in[p+1];
      Bt[l] = (const float*)d_in[p+2]; G[l] = (const float*)d_in[p+3]; p += 4;
    }
    xyz = (const float*)d_in[48];
  }

  // workspace carve-out (~119 MB)
  char* ws = (char*)d_ws; size_t off = 0;
  auto alloc = [&](size_t bytes) -> void* {
    void* p = (void*)(ws + off);
    off = (off + bytes + 255) & ~(size_t)255;
    return p;
  };
  float*     sums  = (float*)alloc(512 * 4);
  float*     sumsq = (float*)alloc(512 * 4);
  float*     nx1   = (float*)alloc((size_t)16 * 512 * 3 * 4);
  float*     nx2   = (float*)alloc((size_t)16 * 128 * 3 * 4);
  int*       idx1  = (int*)  alloc((size_t)16 * 512 * 32 * 4);
  int*       idx2  = (int*)  alloc((size_t)16 * 128 * 64 * 4);
  float*     l1p   = (float*)alloc((size_t)16 * 512 * 128 * 4);
  float*     l2p   = (float*)alloc((size_t)16 * 128 * 256 * 4);
  float*     l3p   = (float*)alloc((size_t)16 * 512 * 4);
  _Float16*  headx = (_Float16*)alloc((size_t)16 * 512 * 2);
  _Float16*  wp    = (_Float16*)alloc((size_t)512 * 512 * 2);
  _Float16*  bufA  = (_Float16*)alloc((size_t)131072 * 160 * 2);  // 40 MB
  _Float16*  bufB  = (_Float16*)alloc((size_t)262144 * 128 * 2);  // 64 MB

  auto run_layer = [&](const _Float16* Xin, int M, int l, _Float16* Yout) {
    int Cout = LC[l], Cin = LI[l];
    int Kpad = (Cin + 31) / 32 * 32;
    int Cpad = Cout; // all BN layers have Cout % 16 == 0
    zero2_kernel<<<1, 512, 0, stream>>>(sums, sumsq, Cpad);
    int wn = Cpad * Kpad;
    cvt_weight<<<cdiv_ll(wn, 256), 256, 0, stream>>>(W[l], Cout, Cin, wp, Kpad, Cpad);
    int tiles = (M / 16) * (Cpad / 16);
    gemm_wmma<<<cdiv_ll(tiles, 8), 256, 0, stream>>>(Xin, wp, Yout, sums, sumsq, M, Kpad, Cpad);
    long long tot = (long long)M * Cpad;
    bn_relu<<<cdiv_ll(tot, 256), 256, 0, stream>>>(Yout, sums, sumsq, G[l], Bt[l], M, Cout, Cpad, tot);
  };

  const int B = 16;
  // ---------------- SA1: N=8192 -> S=512, K=32, r=0.08, 6->64->64->128 ----
  fps_kernel<<<B, 1024, 0, stream>>>(xyz, 8192, 512, nx1);
  ballquery_kernel<<<cdiv_ll(B * 512, 8), 256, 0, stream>>>(xyz, nx1, idx1, B, 8192, 512, 32, 0.08f * 0.08f);
  {
    int total = B * 512 * 32;                     // 262144 rows, Cpad=32
    group_kernel<<<cdiv_ll(total, 256), 256, 0, stream>>>(xyz, xyz, 3, 8192, nx1, idx1, bufA, 512, 32, 32, total);
    run_layer(bufA, total, 0, bufB);              // 6 -> 64
    run_layer(bufB, total, 1, bufA);              // 64 -> 64
    run_layer(bufA, total, 2, bufB);              // 64 -> 128
    maxpool_kernel<<<cdiv_ll((long long)B * 512 * 128, 256), 256, 0, stream>>>(bufB, l1p, B * 512, 32, 128, 128);
  }
  // ---------------- SA2: N=512 -> S=128, K=64, r=0.16, 131->128->128->256 -
  fps_kernel<<<B, 1024, 0, stream>>>(nx1, 512, 128, nx2);
  ballquery_kernel<<<cdiv_ll(B * 128, 8), 256, 0, stream>>>(nx1, nx2, idx2, B, 512, 128, 64, 0.16f * 0.16f);
  {
    int total = B * 128 * 64;                     // 131072 rows, Cpad=160
    group_kernel<<<cdiv_ll(total, 256), 256, 0, stream>>>(nx1, l1p, 128, 512, nx2, idx2, bufA, 128, 64, 160, total);
    run_layer(bufA, total, 3, bufB);              // 131 -> 128
    run_layer(bufB, total, 4, bufA);              // 128 -> 128
    run_layer(bufA, total, 5, bufB);              // 128 -> 256
    maxpool_kernel<<<cdiv_ll((long long)B * 128 * 256, 256), 256, 0, stream>>>(bufB, l2p, B * 128, 64, 256, 256);
  }
  // ---------------- SA3 (group all): 259->256->256->512, max over 128 -----
  {
    int total = B * 128;                          // 2048 rows, Cpad=288
    groupall_kernel<<<cdiv_ll(total, 256), 256, 0, stream>>>(nx2, l2p, 256, 128, bufA, 288, total);
    run_layer(bufA, total, 6, bufB);              // 259 -> 256
    run_layer(bufB, total, 7, bufA);              // 256 -> 256
    run_layer(bufA, total, 8, bufB);              // 256 -> 512
    maxpool_kernel<<<cdiv_ll((long long)B * 512, 256), 256, 0, stream>>>(bufB, l3p, B, 128, 512, 512);
  }
  // ---------------- Head: fc1 (BN), fc2 (BN), fc3 (bias, no BN) -----------
  cvt_f16<<<cdiv_ll(B * 512, 256), 256, 0, stream>>>(l3p, headx, B * 512);
  run_layer(headx, B, 9, bufA);                   // 512 -> 256 (BN over batch)
  run_layer(bufA, B, 10, bufB);                   // 256 -> 128
  fc_naive<<<cdiv_ll(B * 40, 256), 256, 0, stream>>>(bufB, W[11], Bb[11], (float*)d_out, B, 40, 128);
}
